// GATv2Backbone_7576322310466
// MI455X (gfx1250) — matrix operand order repaired and checked by reading, not compiled, not run on hardware
//
#include <hip/hip_runtime.h>
#include <hip/hip_bf16.h>

// ---------------- problem constants (match reference) ----------------
#define NN    25000      // nodes
#define EE    400000     // edges
#define FIN   64
#define HIDC  128
#define HEADS_C 4
#define EDIM  6
#define TV    5
#define MV    4
#define BV    16
#define GRPS  (BV*TV*MV) // 320 pooling groups

typedef __attribute__((ext_vector_type(16))) _Float16 v16h;
typedef __attribute__((ext_vector_type(8)))  float    v8f;

static inline int cdiv(int a, int b) { return (a + b - 1) / b; }

// ---------------- device helpers ----------------
__device__ __forceinline__ float waveRedSum(float v) {
#pragma unroll
  for (int off = 16; off > 0; off >>= 1) v += __shfl_xor(v, off, 32);
  return v;
}

// float atomic-max via int/uint monotonic bit trick (works for mixed signs)
__device__ __forceinline__ void atomicMaxF(float* addr, float val) {
  if (!(val < 0.0f)) atomicMax((int*)addr, __float_as_int(val));
  else               atomicMin((unsigned int*)addr, __float_as_uint(val));
}

// ---------------- generic fill ----------------
__global__ void k_fill(float* p, float v, int n) {
  int i = blockIdx.x * blockDim.x + threadIdx.x;
  if (i < n) p[i] = v;
}

// ---------------- weight prep: fp32 [K,Nc] row-major -> f16 [Nc,K] (col-major B) ----------------
__global__ void k_prep_weight(const float* __restrict__ W, _Float16* __restrict__ Wt,
                              int K, int Nc) {
  int idx = blockIdx.x * blockDim.x + threadIdx.x;
  if (idx < K * Nc) {
    int n = idx / K, k = idx % K;
    Wt[idx] = (_Float16)W[k * Nc + n];
  }
}

// ---------------- WMMA GEMM: C[M,Nc] = act(A[M,K] @ B[K,Nc] + bias) ----------------
// A fp32 row-major (converted to f16 in-flight), Bt f16 stored [Nc,K] (B^T, contiguous K).
// One wave computes a 16x64 strip (4 WMMA tiles along N). K is a compile-time constant
// (multiple of 32); Nc must be a multiple of 64 (grid.y = Nc/64).
// Row index is CLAMPED (not zero-filled) so A loads are branchless float4s; rows >= M
// compute duplicated-row garbage that is masked off at the store.
// ACT: 0 = none, 1 = relu, 2 = tanh
template<int K, int ACT>
__global__ void k_wmma_gemm(const float* __restrict__ A, const _Float16* __restrict__ Bt,
                            const float* __restrict__ bias, float* __restrict__ C,
                            int M, int Nc) {
  const int lane = threadIdx.x & 31;
  const int g    = lane >> 4;   // half-wave selector
  const int l15  = lane & 15;
  int rowL = blockIdx.x * 16 + l15;
  rowL = rowL < M ? rowL : M - 1;           // clamp: branchless, always-valid loads
  const int colBase = blockIdx.y * 64;
  const float4*   Arow4 = (const float4*)(A + (size_t)rowL * K);   // 256B-aligned rows
  const _Float16* B0    = Bt + (size_t)(colBase + l15) * K;

  v8f acc[4] = {};
#pragma unroll
  for (int k0 = 0; k0 < K; k0 += 32) {
    // A 16x32 f16 layout: lane L -> row L&15; elem e: K=k0+16g+e
    const int base = (k0 + g * 16) >> 2;    // float4 index
    float4 a0 = Arow4[base],     a1 = Arow4[base + 1];
    float4 a2 = Arow4[base + 2], a3 = Arow4[base + 3];
    v16h a;
    a[0]  = (_Float16)a0.x; a[1]  = (_Float16)a0.y; a[2]  = (_Float16)a0.z; a[3]  = (_Float16)a0.w;
    a[4]  = (_Float16)a1.x; a[5]  = (_Float16)a1.y; a[6]  = (_Float16)a1.z; a[7]  = (_Float16)a1.w;
    a[8]  = (_Float16)a2.x; a[9]  = (_Float16)a2.y; a[10] = (_Float16)a2.z; a[11] = (_Float16)a2.w;
    a[12] = (_Float16)a3.x; a[13] = (_Float16)a3.y; a[14] = (_Float16)a3.z; a[15] = (_Float16)a3.w;
    // B 32x16 f16 (stored as B^T): lane L -> col l15; elem e: K = k0+16g+e (contiguous 32B)
    const int kb = k0 + g * 16;
#pragma unroll
    for (int t = 0; t < 4; ++t) {
      v16h b = *(const v16h*)(B0 + (size_t)t * 16 * K + kb);
      acc[t] = __builtin_amdgcn_wmma_f32_16x16x32_f16(
          /*neg_a=*/false, a, /*neg_b=*/false, b,
          /*c_mod=*/(short)0, acc[t], /*reuse_a=*/false, /*reuse_b=*/false);
    }
  }

#pragma unroll
  for (int t = 0; t < 4; ++t) {
    const float bv = bias[colBase + t * 16 + l15];
#pragma unroll
    for (int r = 0; r < 8; ++r) {           // D: VGPR r -> row r+8g, col l15
      int m = blockIdx.x * 16 + r + 8 * g;
      if (m < M) {
        float v = acc[t][r] + bv;
        if (ACT == 1)      v = fmaxf(v, 0.0f);
        else if (ACT == 2) v = tanhf(v);
        C[(size_t)m * Nc + colBase + t * 16 + l15] = v;
      }
    }
  }
}

// ---------------- LayerNorm over 128 features (one wave per row) ----------------
// act: 0 = none, 1 = silu
__global__ void k_ln(const float* __restrict__ in, const float* __restrict__ pre_bias,
                     const float* __restrict__ gam, const float* __restrict__ bet,
                     float* __restrict__ out, int N, int act) {
  int row  = blockIdx.x * (blockDim.x >> 5) + (threadIdx.x >> 5);
  int lane = threadIdx.x & 31;
  if (row >= N) return;
  float v[4]; float sum = 0.0f;
#pragma unroll
  for (int i = 0; i < 4; ++i) {
    int f = i * 32 + lane;
    v[i] = in[(size_t)row * HIDC + f] + (pre_bias ? pre_bias[f] : 0.0f);
    sum += v[i];
  }
  sum = waveRedSum(sum);
  float mu = sum * (1.0f / HIDC);
  float var = 0.0f;
#pragma unroll
  for (int i = 0; i < 4; ++i) { float d = v[i] - mu; var += d * d; }
  var = waveRedSum(var) * (1.0f / HIDC);
  float inv = rsqrtf(var + 1e-5f);
#pragma unroll
  for (int i = 0; i < 4; ++i) {
    int f = i * 32 + lane;
    float o = (v[i] - mu) * inv * gam[f] + bet[f];
    if (act == 1) o = o / (1.0f + __expf(-o));   // silu
    out[(size_t)row * HIDC + f] = o;
  }
}

// ---------------- self-loop attr: mean of incoming edge attrs ----------------
__global__ void k_loop_accum(const int* __restrict__ dst, const float* __restrict__ eattr,
                             float* __restrict__ attr_sum, float* __restrict__ cnt) {
  int e = blockIdx.x * blockDim.x + threadIdx.x;
  if (e >= EE) return;
  int d = dst[e];
  atomicAdd(&cnt[d], 1.0f);
#pragma unroll
  for (int j = 0; j < EDIM; ++j)
    atomicAdd(&attr_sum[(size_t)d * EDIM + j], eattr[(size_t)e * EDIM + j]);
}
__global__ void k_loop_final(const float* __restrict__ attr_sum, const float* __restrict__ cnt,
                             float* __restrict__ loop_attr) {
  int n = blockIdx.x * blockDim.x + threadIdx.x;
  if (n >= NN) return;
  float c = fmaxf(cnt[n], 1.0f);
#pragma unroll
  for (int j = 0; j < EDIM; ++j)
    loop_attr[(size_t)n * EDIM + j] = attr_sum[(size_t)n * EDIM + j] / c;
}

// ---------------- GATv2 edge pass 1: logits + segment max (wave per edge) ----------------
__global__ void k_edge_p1(const int* __restrict__ src, const int* __restrict__ dst,
                          const float* __restrict__ eattr, const float* __restrict__ loop_attr,
                          const float* __restrict__ xl, const float* __restrict__ xr,
                          const float* __restrict__ We, const float* __restrict__ att,
                          float* __restrict__ logits, float* __restrict__ lmax) {
  int w    = (blockIdx.x * blockDim.x + threadIdx.x) >> 5;
  int lane = threadIdx.x & 31;
  if (w >= EE + NN) return;
  int s, d; const float* ea;
  if (w < EE) { s = src[w]; d = dst[w]; ea = eattr + (size_t)w * EDIM; }
  else        { s = d = w - EE;         ea = loop_attr + (size_t)(w - EE) * EDIM; }
  float e0 = ea[0], e1 = ea[1], e2 = ea[2], e3 = ea[3], e4 = ea[4], e5 = ea[5];
  float c[HEADS_C];
#pragma unroll
  for (int h = 0; h < HEADS_C; ++h) {
    int f = h * 32 + lane;
    const float* Wf = We + f;
    float ee = e0 * Wf[0] + e1 * Wf[HIDC] + e2 * Wf[2 * HIDC] +
               e3 * Wf[3 * HIDC] + e4 * Wf[4 * HIDC] + e5 * Wf[5 * HIDC];
    float m = xl[(size_t)s * HIDC + f] + xr[(size_t)d * HIDC + f] + ee;
    m = m > 0.0f ? m : 0.2f * m;          // leaky_relu(0.2)
    c[h] = m * att[h * 32 + lane];
  }
#pragma unroll
  for (int h = 0; h < HEADS_C; ++h) c[h] = waveRedSum(c[h]);
  if (lane == 0) {
#pragma unroll
    for (int h = 0; h < HEADS_C; ++h) {
      logits[(size_t)w * HEADS_C + h] = c[h];
      atomicMaxF(&lmax[(size_t)d * HEADS_C + h], c[h]);
    }
  }
}

// ---------------- pass 2: exp + segment sum ----------------
__global__ void k_edge_p2(const int* __restrict__ dst, float* __restrict__ logits,
                          const float* __restrict__ lmax, float* __restrict__ den) {
  int e = blockIdx.x * blockDim.x + threadIdx.x;
  if (e >= EE + NN) return;
  int d = (e < EE) ? dst[e] : e - EE;
#pragma unroll
  for (int h = 0; h < HEADS_C; ++h) {
    float ex = __expf(logits[(size_t)e * HEADS_C + h] - lmax[(size_t)d * HEADS_C + h]);
    logits[(size_t)e * HEADS_C + h] = ex;
    atomicAdd(&den[(size_t)d * HEADS_C + h], ex);
  }
}

// ---------------- pass 3: weighted scatter-add (wave per edge) ----------------
__global__ void k_edge_p3(const int* __restrict__ src, const int* __restrict__ dst,
                          const float* __restrict__ xl, const float* __restrict__ ex,
                          const float* __restrict__ den, float* __restrict__ out) {
  int w    = (blockIdx.x * blockDim.x + threadIdx.x) >> 5;
  int lane = threadIdx.x & 31;
  if (w >= EE + NN) return;
  int s, d;
  if (w < EE) { s = src[w]; d = dst[w]; }
  else        { s = d = w - EE; }
#pragma unroll
  for (int h = 0; h < HEADS_C; ++h) {
    int f = h * 32 + lane;
    float alpha = ex[(size_t)w * HEADS_C + h] / den[(size_t)d * HEADS_C + h];
    atomicAdd(&out[(size_t)d * HIDC + f], alpha * xl[(size_t)s * HIDC + f]);
  }
}

// ---------------- pooling: node scores s = tanh(h@W1+b1)@W2+b2 (W1 GEMM done via WMMA) ----
__global__ void k_score(const float* __restrict__ ptmp, const float* __restrict__ W2,
                        const float* __restrict__ b2, float* __restrict__ s) {
  int n = blockIdx.x * blockDim.x + threadIdx.x;
  if (n >= NN) return;
  float acc = 0.0f;
#pragma unroll
  for (int j = 0; j < 64; ++j) acc += ptmp[(size_t)n * 64 + j] * W2[j];
  s[n] = acc + b2[0];
}

__global__ void k_pool_p1(const float* __restrict__ s, const int* __restrict__ batch,
                          const int* __restrict__ ntype, const int* __restrict__ tmask,
                          int* __restrict__ gidx, float* __restrict__ smax) {
  int n = blockIdx.x * blockDim.x + threadIdx.x;
  if (n >= NN) return;
  int g = batch[n] * (TV * MV) + tmask[n] * MV + ntype[n];
  gidx[n] = g;
  atomicMaxF(&smax[g], s[n]);
}
__global__ void k_pool_p2(const float* __restrict__ s, const int* __restrict__ gidx,
                          const float* __restrict__ smax, float* __restrict__ exn,
                          float* __restrict__ den) {
  int n = blockIdx.x * blockDim.x + threadIdx.x;
  if (n >= NN) return;
  float ex = __expf(s[n] - smax[gidx[n]]);
  exn[n] = ex;
  atomicAdd(&den[gidx[n]], ex);
}
__global__ void k_pool_p3(const float* __restrict__ h, const int* __restrict__ gidx,
                          const float* __restrict__ exn, const float* __restrict__ den,
                          float* __restrict__ pooled) {
  int idx = blockIdx.x * blockDim.x + threadIdx.x;
  if (idx >= NN * HIDC) return;
  int n = idx >> 7, f = idx & 127;
  int g = gidx[n];
  float w = exn[n] / den[g];
  atomicAdd(&pooled[(size_t)g * HIDC + f], h[(size_t)n * HIDC + f] * w);
}

// ---------------- gated fusion over node types (one block per (b,t)) ----------------
__global__ void k_gate(const float* __restrict__ pooled, const float* __restrict__ gW,
                       const float* __restrict__ gb, float* __restrict__ out) {
  int bt = blockIdx.x;             // 0..B*T-1
  int tid = threadIdx.x;           // 0..127
  int m = tid >> 5, lane = tid & 31;
  __shared__ float sm[MV];
  const float* pt = pooled + ((size_t)bt * MV + m) * HIDC;
  float acc = 0.0f;
#pragma unroll
  for (int i = 0; i < 4; ++i) { int f = i * 32 + lane; acc += pt[f] * gW[f]; }
  acc = waveRedSum(acc);
  if (lane == 0) sm[m] = acc + gb[0];
  __syncthreads();
  float g0 = sm[0], g1 = sm[1], g2 = sm[2], g3 = sm[3];
  float mx = fmaxf(fmaxf(g0, g1), fmaxf(g2, g3));
  float w0 = __expf(g0 - mx), w1 = __expf(g1 - mx), w2 = __expf(g2 - mx), w3 = __expf(g3 - mx);
  float inv = 1.0f / (w0 + w1 + w2 + w3);
  int f = tid;
  const float* p = pooled + (size_t)bt * MV * HIDC;
  out[(size_t)bt * HIDC + f] =
      (p[f] * w0 + p[HIDC + f] * w1 + p[2 * HIDC + f] * w2 + p[3 * HIDC + f] * w3) * inv;
}

// ============================ host launcher ============================
extern "C" void kernel_launch(void* const* d_in, const int* in_sizes, int n_in,
                              void* d_out, int out_size, void* d_ws, size_t ws_size,
                              hipStream_t stream) {
  // ---- input mapping (setup_inputs insertion order; params flattened depth-first
  //      in insertion order: enc_W1,enc_b1,enc_ln_g,enc_ln_b,enc_W2,enc_b2,
  //      pool_W1,pool_b1,pool_W2,pool_b2,gate_W,gate_b, convs[l]{Wl,bl,Wr,br,We,att,bias}, norms[l]{g,b})
  const float* x        = (const float*)d_in[0];
  const int*   eidx     = (const int*)d_in[1];
  const int*   srcI     = eidx;
  const int*   dstI     = eidx + EE;
  const float* eattr    = (const float*)d_in[2];
  const int*   batchI   = (const int*)d_in[3];
  const int*   ntypeI   = (const int*)d_in[4];
  const int*   tmaskI   = (const int*)d_in[5];
  // d_in[6..8] = T, M, B scalars (compile-time constants here)
  const float* enc_W1   = (const float*)d_in[9];
  const float* enc_b1   = (const float*)d_in[10];
  const float* enc_ln_g = (const float*)d_in[11];
  const float* enc_ln_b = (const float*)d_in[12];
  const float* enc_W2   = (const float*)d_in[13];
  const float* enc_b2   = (const float*)d_in[14];
  const float* pool_W1  = (const float*)d_in[15];
  const float* pool_b1  = (const float*)d_in[16];
  const float* pool_W2  = (const float*)d_in[17];
  const float* pool_b2  = (const float*)d_in[18];
  const float* gate_W   = (const float*)d_in[19];
  const float* gate_b   = (const float*)d_in[20];
  const float* convW[2][7];   // Wl, bl, Wr, br, We, att, bias
  const float* normW[2][2];   // g, b
  for (int l = 0; l < 2; ++l) {
    for (int j = 0; j < 7; ++j) convW[l][j] = (const float*)d_in[21 + l * 7 + j];
    for (int j = 0; j < 2; ++j) normW[l][j] = (const float*)d_in[35 + l * 2 + j];
  }
  float* out = (float*)d_out;   // [B,T,HID] = [16,5,128]

  // ---- workspace carve (floats) ----
  float* ws = (float*)d_ws;
  size_t off = 0;
  auto take = [&](size_t n) { float* p = ws + off; off += n; return p; };
  float* hA       = take((size_t)NN * HIDC);
  float* hB       = take((size_t)NN * HIDC);
  float* xl       = take((size_t)NN * HIDC);
  float* xr       = take((size_t)NN * HIDC);
  float* outb     = take((size_t)NN * HIDC);
  float* logits   = take((size_t)(EE + NN) * HEADS_C);
  float* lmax     = take((size_t)NN * HEADS_C);
  float* den      = take((size_t)NN * HEADS_C);
  float* attr_sum = take((size_t)NN * EDIM);
  float* cntb     = take((size_t)NN);
  float* loopat   = take((size_t)NN * EDIM);
  float* ptmp     = take((size_t)NN * 64);
  float* sbuf     = take((size_t)NN);
  float* exn      = take((size_t)NN);
  float* smax     = take((size_t)GRPS);
  float* denp     = take((size_t)GRPS);
  float* pooled   = take((size_t)GRPS * HIDC);
  int*   gidx     = (int*)take((size_t)NN);
  _Float16* f16base = (_Float16*)take(64 * 1024);   // 128K halfs region
  _Float16* encW1t = f16base;                 // [128,64]
  _Float16* encW2t = encW1t + 64 * HIDC;      // [128,128]
  _Float16* Wlt[2], *Wrt[2];
  Wlt[0] = encW2t + HIDC * HIDC;
  Wrt[0] = Wlt[0] + HIDC * HIDC;
  Wlt[1] = Wrt[0] + HIDC * HIDC;
  Wrt[1] = Wlt[1] + HIDC * HIDC;
  _Float16* pW1t = Wrt[1] + HIDC * HIDC;      // [64,128]
  (void)ws_size; (void)n_in; (void)in_sizes; (void)out_size;

  const int TPB = 256;

  // ---- 1. weight prep (fp32 -> f16, B^T layout) ----
  k_prep_weight<<<cdiv(FIN * HIDC, TPB), TPB, 0, stream>>>(enc_W1, encW1t, FIN, HIDC);
  k_prep_weight<<<cdiv(HIDC * HIDC, TPB), TPB, 0, stream>>>(enc_W2, encW2t, HIDC, HIDC);
  for (int l = 0; l < 2; ++l) {
    k_prep_weight<<<cdiv(HIDC * HIDC, TPB), TPB, 0, stream>>>(convW[l][0], Wlt[l], HIDC, HIDC);
    k_prep_weight<<<cdiv(HIDC * HIDC, TPB), TPB, 0, stream>>>(convW[l][2], Wrt[l], HIDC, HIDC);
  }
  k_prep_weight<<<cdiv(HIDC * 64, TPB), TPB, 0, stream>>>(pool_W1, pW1t, HIDC, 64);

  // ---- 2. self-loop attrs (mean of incoming) ----
  k_fill<<<cdiv(NN * EDIM, TPB), TPB, 0, stream>>>(attr_sum, 0.0f, NN * EDIM);
  k_fill<<<cdiv(NN, TPB), TPB, 0, stream>>>(cntb, 0.0f, NN);
  k_loop_accum<<<cdiv(EE, TPB), TPB, 0, stream>>>(dstI, eattr, attr_sum, cntb);
  k_loop_final<<<cdiv(NN, TPB), TPB, 0, stream>>>(attr_sum, cntb, loopat);

  // ---- 3. node encoder: relu(x@W1+b1) -> LN -> @W2+b2 ----
  {
    dim3 g1(cdiv(NN, 16), HIDC / 64);
    k_wmma_gemm<FIN, 1><<<g1, 32, 0, stream>>>(x, encW1t, enc_b1, hA, NN, HIDC);
    k_ln<<<cdiv(NN, 8), 256, 0, stream>>>(hA, nullptr, enc_ln_g, enc_ln_b, hA, NN, 0);
    k_wmma_gemm<HIDC, 0><<<g1, 32, 0, stream>>>(hA, encW2t, enc_b2, hB, NN, HIDC);
  }

  // ---- 4. GATv2 layers ----
  const int nWaveBlocks = cdiv((EE + NN) * 32, TPB);
  for (int l = 0; l < 2; ++l) {
    dim3 g1(cdiv(NN, 16), HIDC / 64);
    k_wmma_gemm<HIDC, 0><<<g1, 32, 0, stream>>>(hB, Wlt[l], convW[l][1], xl, NN, HIDC);
    k_wmma_gemm<HIDC, 0><<<g1, 32, 0, stream>>>(hB, Wrt[l], convW[l][3], xr, NN, HIDC);
    k_fill<<<cdiv(NN * HEADS_C, TPB), TPB, 0, stream>>>(lmax, -1e30f, NN * HEADS_C);
    k_fill<<<cdiv(NN * HEADS_C, TPB), TPB, 0, stream>>>(den, 0.0f, NN * HEADS_C);
    k_fill<<<cdiv(NN * HIDC, TPB), TPB, 0, stream>>>(outb, 0.0f, NN * HIDC);
    k_edge_p1<<<nWaveBlocks, TPB, 0, stream>>>(srcI, dstI, eattr, loopat, xl, xr,
                                               convW[l][4], convW[l][5], logits, lmax);
    k_edge_p2<<<cdiv(EE + NN, TPB), TPB, 0, stream>>>(dstI, logits, lmax, den);
    k_edge_p3<<<nWaveBlocks, TPB, 0, stream>>>(srcI, dstI, xl, logits, den, outb);
    // h = silu(LN(out + conv_bias))
    k_ln<<<cdiv(NN, 8), 256, 0, stream>>>(outb, convW[l][6], normW[l][0], normW[l][1], hB, NN, /*silu*/1);
  }

  // ---- 5. attention pooling over (batch,time,type) groups ----
  {
    dim3 gp(cdiv(NN, 16), 64 / 64);
    k_wmma_gemm<HIDC, 2><<<gp, 32, 0, stream>>>(hB, pW1t, pool_b1, ptmp, NN, 64);
    k_score<<<cdiv(NN, TPB), TPB, 0, stream>>>(ptmp, pool_W2, pool_b2, sbuf);
    k_fill<<<cdiv(GRPS, TPB), TPB, 0, stream>>>(smax, -1e30f, GRPS);
    k_fill<<<cdiv(GRPS, TPB), TPB, 0, stream>>>(denp, 0.0f, GRPS);
    k_fill<<<cdiv(GRPS * HIDC, TPB), TPB, 0, stream>>>(pooled, 0.0f, GRPS * HIDC);
    k_pool_p1<<<cdiv(NN, TPB), TPB, 0, stream>>>(sbuf, batchI, ntypeI, tmaskI, gidx, smax);
    k_pool_p2<<<cdiv(NN, TPB), TPB, 0, stream>>>(sbuf, gidx, smax, exn, denp);
    k_pool_p3<<<cdiv(NN * HIDC, TPB), TPB, 0, stream>>>(hB, gidx, exn, denp, pooled);
  }

  // ---- 6. gated fusion across node types -> [B,T,HID] ----
  k_gate<<<BV * TV, HIDC, 0, stream>>>(pooled, gate_W, gate_b, out);
}
